// SelfAttention_29875792511159
// MI455X (gfx1250) — matrix operand order repaired and checked by reading, not compiled, not run on hardware
//
#include <hip/hip_runtime.h>

typedef __attribute__((ext_vector_type(16))) _Float16 v16h;
typedef __attribute__((ext_vector_type(8)))  float    v8f;

#define H_HEADS 8
#define DK 64
#define SEQ 4096
#define IN_DIM 512
#define BATCH 2

__device__ __forceinline__ v8f wmma16(v16h a, v16h b, v8f c) {
  return __builtin_amdgcn_wmma_f32_16x16x32_f16(false, a, false, b, (short)0, c,
                                                false, false);
}

// WT[c*512 + r] = (f16) W[r*512 + c]   (transpose + downconvert, 512x512)
__global__ void convert_wt(const float* __restrict__ W, _Float16* __restrict__ WT) {
  int t = blockIdx.x * blockDim.x + threadIdx.x;  // 0..262143
  int r = t >> 9, c = t & 511;
  WT[(size_t)c * IN_DIM + r] = (_Float16)W[t];
}

// out = x[8192,512] @ W[512,512], W provided as transposed f16 WT (row = out col).
// dst layout: TR==false -> [B,H,N,64]; TR==true -> [B,H,64,N]
template <bool TR>
__global__ void qkv_gemm(const float* __restrict__ x, const _Float16* __restrict__ WT,
                         _Float16* __restrict__ dst) {
  const int wid = threadIdx.x >> 5;
  const int lane = threadIdx.x & 31;
  const int lo = lane & 15, hi = lane >> 4;
  const int gw = blockIdx.x * 8 + wid;  // 0..4095
  const int mt = gw >> 3;               // 16-row tile, 0..511
  const int ns = gw & 7;                // head / 64-col strip
  const int row = mt * 16 + lo;

  v8f acc[4] = {};
  for (int k0 = 0; k0 < IN_DIM; k0 += 32) {
    // A fragment (16x32 f16): lane row = lo, K = e + 8*hi (e<8), e+8+8*hi (e>=8)
    const float* asrc = x + (size_t)row * IN_DIM + k0 + 8 * hi;
    v16h a;
#pragma unroll
    for (int e = 0; e < 8; ++e) a[e] = (_Float16)asrc[e];
#pragma unroll
    for (int e = 8; e < 16; ++e) a[e] = (_Float16)asrc[e + 8];
#pragma unroll
    for (int j = 0; j < 4; ++j) {
      // B fragment (32x16 f16): lane col = lo, K = e + 8*hi (e<8), e+8+8*hi (e>=8)
      const _Float16* bsrc =
          WT + (size_t)(ns * 64 + j * 16 + lo) * IN_DIM + k0 + 8 * hi;
      v16h bf;
      *((uint4*)&bf) = *(const uint4*)bsrc;
      *((uint4*)&bf + 1) = *(const uint4*)(bsrc + 16);
      acc[j] = wmma16(a, bf, acc[j]);
    }
  }
  const int b = (mt * 16) >> 12;  // 16-row tile never straddles batch boundary
#pragma unroll
  for (int j = 0; j < 4; ++j) {
#pragma unroll
    for (int r = 0; r < 8; ++r) {
      int M = mt * 16 + r + 8 * hi;  // C/D layout: row = r + 8*hi, col = lane&15
      int n = M & (SEQ - 1);
      int d = j * 16 + lo;
      size_t idx;
      if (TR)
        idx = ((size_t)((b * H_HEADS + ns) * DK + d)) * SEQ + n;
      else
        idx = ((size_t)((b * H_HEADS + ns) * SEQ + n)) * DK + d;
      dst[idx] = (_Float16)acc[j][r];
    }
  }
}

// Flash attention, transposed orientation.
// One wave owns 16 output rows n (= lanes); streams m in steps of 64.
// S^T = Q . (K/8)^T  (M=m, N=n)  -> softmax dim m lives in regs within a lane.
// O^T = V^T . P^T    (M=d, N=n).
// Kh,Qh: [B,H,N,64] f16; Vt: [B,H,64,N] f16; out: [B,N,512] f32
#define PSTR 72  // f16 row stride for P tile (padding kills bank conflicts)
#define OSTR 68  // f32 row stride for O tile
__global__ void flash_attn(const _Float16* __restrict__ Kh,
                           const _Float16* __restrict__ Qh,
                           const _Float16* __restrict__ Vt, float* __restrict__ out) {
  __shared__ _Float16 plds[8][16 * PSTR];  // per-wave P^T relayout tile (n x m)
  __shared__ float    olds[8][16 * OSTR];  // per-wave O transpose tile  (n x d)
  const int wid = threadIdx.x >> 5;
  const int lane = threadIdx.x & 31;
  const int lo = lane & 15, hi = lane >> 4;
  const int gw = blockIdx.x * 8 + wid;  // 0..4095
  const int b = gw >> 11;
  const int h = (gw >> 8) & 7;
  const int nt = gw & 255;
  const int n0 = nt * 16;

  const size_t bh = (size_t)(b * H_HEADS + h);
  const _Float16* Kp = Kh + bh * SEQ * DK;
  const _Float16* Qp = Qh + bh * SEQ * DK;
  const _Float16* Vp = Vt + bh * (size_t)DK * SEQ;

  // B fragments of (K/8)^T for this wave's 16 n-columns; d chunks [0,32),[32,64).
  // B layout: lane col n = lo, element e -> d = e + 8*hi (e<8), e + 8 + 8*hi (e>=8).
  v16h bk0, bk1;
  {
    const _Float16* ks = Kp + (size_t)(n0 + lo) * DK + 8 * hi;
    *((uint4*)&bk0) = *(const uint4*)ks;
    *((uint4*)&bk0 + 1) = *(const uint4*)(ks + 16);
    *((uint4*)&bk1) = *(const uint4*)(ks + 32);
    *((uint4*)&bk1 + 1) = *(const uint4*)(ks + 48);
#pragma unroll
    for (int e = 0; e < 16; ++e) {  // 1/sqrt(64), exact in f16
      bk0[e] = bk0[e] * (_Float16)0.125f;
      bk1[e] = bk1[e] * (_Float16)0.125f;
    }
  }

  v8f o[4] = {};               // O^T: frag j reg r -> d = j*16+r+8*hi, n = lo
  float mi = -__builtin_inff();  // per-lane (per n) running max / sum
  float li = 0.f;

  for (int m0 = 0; m0 < SEQ; m0 += 64) {
    // ---- S^T: 4 fragments along m; A = Q rows (m), K-dim = d in 2 chunks ----
    v8f s[4] = {};
#pragma unroll
    for (int c = 0; c < 4; ++c) {
      const _Float16* qs = Qp + (size_t)(m0 + c * 16 + lo) * DK + 8 * hi;
      v16h a0, a1;
      *((uint4*)&a0) = *(const uint4*)qs;
      *((uint4*)&a0 + 1) = *(const uint4*)(qs + 16);
      *((uint4*)&a1) = *(const uint4*)(qs + 32);
      *((uint4*)&a1 + 1) = *(const uint4*)(qs + 48);
      s[c] = wmma16(a0, bk0, s[c]);
      s[c] = wmma16(a1, bk1, s[c]);
    }
    // ---- online softmax over m: in-lane tree + one cross-half combine ----
    float tm = -__builtin_inff();
#pragma unroll
    for (int c = 0; c < 4; ++c)
#pragma unroll
      for (int r = 0; r < 8; ++r) tm = fmaxf(tm, s[c][r]);
    tm = fmaxf(tm, __shfl_xor(tm, 16, 32));
    float nm = fmaxf(mi, tm);
    float alpha = __expf(mi - nm);
    float rs = 0.f;
#pragma unroll
    for (int c = 0; c < 4; ++c)
#pragma unroll
      for (int r = 0; r < 8; ++r) {
        float p = __expf(s[c][r] - nm);
        rs += p;
        // P^T value at (m_local = c*16+r+8*hi, n = lo) -> plds[n][m_local]
        plds[wid][lo * PSTR + c * 16 + r + 8 * hi] = (_Float16)p;
      }
    rs += __shfl_xor(rs, 16, 32);
    li = li * alpha + rs;
    mi = nm;
#pragma unroll
    for (int j = 0; j < 4; ++j)
#pragma unroll
      for (int r = 0; r < 8; ++r) o[j][r] *= alpha;

    // ---- P^T as B fragments (m chunks [0,32),[32,64)); reused across 4 d-tiles ----
    v16h pb0, pb1;
    {
      const _Float16* ls = &plds[wid][lo * PSTR + 8 * hi];
      *((uint4*)&pb0) = *(const uint4*)ls;
      *((uint4*)&pb0 + 1) = *(const uint4*)(ls + 16);
      *((uint4*)&pb1) = *(const uint4*)(ls + 32);
      *((uint4*)&pb1 + 1) = *(const uint4*)(ls + 48);
    }
    // ---- O^T += V^T . P^T ; A = V^T rows d (contiguous from transposed V) ----
#pragma unroll
    for (int j = 0; j < 4; ++j) {
      const _Float16* vs = Vp + (size_t)(j * 16 + lo) * SEQ + m0 + 8 * hi;
      v16h a0, a1;
      *((uint4*)&a0) = *(const uint4*)vs;
      *((uint4*)&a0 + 1) = *(const uint4*)(vs + 16);
      *((uint4*)&a1) = *(const uint4*)(vs + 32);
      *((uint4*)&a1 + 1) = *(const uint4*)(vs + 48);
      o[j] = wmma16(a0, pb0, o[j]);
      o[j] = wmma16(a1, pb1, o[j]);
    }
  }

  // ---- epilogue: normalize, transpose O^T through LDS, coalesced f32 stores ----
  const float inv_li = 1.f / li;
#pragma unroll
  for (int j = 0; j < 4; ++j)
#pragma unroll
    for (int r = 0; r < 8; ++r)
      olds[wid][lo * OSTR + j * 16 + r + 8 * hi] = o[j][r] * inv_li;

  const size_t obase = (size_t)(b * SEQ + n0) * (H_HEADS * DK) + h * DK;
#pragma unroll
  for (int it = 0; it < 32; ++it) {
    int flat = it * 32 + lane;      // 16 rows x 64 cols
    int nl = flat >> 6, d = flat & 63;
    out[obase + (size_t)nl * (H_HEADS * DK) + d] = olds[wid][nl * OSTR + d];
  }
}

extern "C" void kernel_launch(void* const* d_in, const int* in_sizes, int n_in,
                              void* d_out, int out_size, void* d_ws, size_t ws_size,
                              hipStream_t stream) {
  (void)in_sizes; (void)n_in; (void)out_size; (void)ws_size;
  const float* x  = (const float*)d_in[0];
  const float* Wk = (const float*)d_in[1];
  const float* Wq = (const float*)d_in[2];
  const float* Wv = (const float*)d_in[3];
  float* out = (float*)d_out;

  _Float16* ws = (_Float16*)d_ws;
  _Float16* WkT = ws;                 //   512*512
  _Float16* WqT = WkT + 262144;
  _Float16* WvT = WqT + 262144;
  _Float16* Khp = WvT + 262144;       // [B,H,N,64]  = 4194304 f16
  _Float16* Qhp = Khp + 4194304;
  _Float16* Vtp = Qhp + 4194304;      // [B,H,64,N]
  // total = 13,369,344 f16 = ~25.5 MB of d_ws

  convert_wt<<<1024, 256, 0, stream>>>(Wk, WkT);
  convert_wt<<<1024, 256, 0, stream>>>(Wq, WqT);
  convert_wt<<<1024, 256, 0, stream>>>(Wv, WvT);

  qkv_gemm<false><<<512, 256, 0, stream>>>(x, WkT, Khp);
  qkv_gemm<false><<<512, 256, 0, stream>>>(x, WqT, Qhp);
  qkv_gemm<true><<<512, 256, 0, stream>>>(x, WvT, Vtp);

  flash_attn<<<512, 256, 0, stream>>>(Khp, Qhp, Vtp, out);
}
